// KANLayer_82575041233195
// MI455X (gfx1250) — matrix operand order repaired and checked by reading, not compiled, not run on hardware
//
#include <hip/hip_runtime.h>

typedef _Float16 v16h __attribute__((ext_vector_type(16)));
typedef float    v8f  __attribute__((ext_vector_type(8)));
typedef __fp16   pk2  __attribute__((ext_vector_type(2)));   // cvt_pkrtz result type

#define B_PTS     131072
#define F_IN      64
#define S_MODES   16
#define O_OUT     32
#define N_CHUNK   32              // K = F*S = 1024 = 32 chunks of 32
#define TILE_M    16
#define N_TILES   (B_PTS / TILE_M)  // 8192
#define THREADS   256
#define WAVES_PB  (THREADS / 32)
#define GRID_B    256
#define LDSK_ELEMS (N_CHUNK * 2 * 32 * 16)   // 32768 halves = 64 KB

// Permutation of the contraction index within a chunk (agreed by A and B):
//   koff in [0,32): f = 2c + ((koff>>3)&1),  s = ((koff&16)>>1) | (koff&7)
// With the hw 16-bit A layout (lane<16: koff = (h&8)*2 + (h&7);
// lane>=16: koff = (h&8)*2 + 8 + (h&7)), half h of each lane maps to
// s == h of the single feature f = 2c + laneGroup.  => one Chebyshev
// chain per lane, packed in order, zero shuffles.

__global__ __launch_bounds__(THREADS) void kan_cheb_wmma(
    const float* __restrict__ x,      // [B_PTS, F_IN]
    const float* __restrict__ kernel, // [F_IN, S_MODES, O_OUT]
    float* __restrict__ out)          // [B_PTS, O_OUT]
{
    __shared__ _Float16 ldsK[LDSK_ELEMS];

    const int tid = threadIdx.x;

    // ---- stage permuted f16 kernel into LDS (B-fragment layout) ----
    // linear i = ((c*2 + t)*32 + l)*16 + h ; B half h of lane l carries
    // koff = (h&8)*2 + (l>=16 ? 8 : 0) + (h&7) ; column o = (l&15) + 16*t
    for (int i = tid; i < LDSK_ELEMS; i += THREADS) {
        int h = i & 15;
        int l = (i >> 4) & 31;
        int t = (i >> 9) & 1;
        int c = i >> 10;
        int koff = (h & 8) * 2 + ((l >= 16) ? 8 : 0) + (h & 7);
        int f = 2 * c + ((koff >> 3) & 1);
        int s = ((koff & 16) >> 1) | (koff & 7);
        int o = (l & 15) + 16 * t;
        ldsK[i] = (_Float16)kernel[(f * S_MODES + s) * O_OUT + o];
    }
    __syncthreads();

    const int lane   = tid & 31;
    const int waveId = tid >> 5;
    const int g      = lane >> 4;      // lane group -> which of the 2 features
    const int mrow   = lane & 15;      // A-matrix row index
    const int totalWaves = GRID_B * WAVES_PB;
    const int wave = blockIdx.x * WAVES_PB + waveId;

    for (int tile = wave; tile < N_TILES; tile += totalWaves) {
        const int r0 = tile * TILE_M;
        const float* __restrict__ xr = x + (size_t)(r0 + mrow) * F_IN;

        v8f acc0 = {};
        v8f acc1 = {};

        #pragma unroll 4
        for (int c = 0; c < N_CHUNK; ++c) {
            // This lane's feature value for this chunk
            float xv   = xr[2 * c + g];
            float twox = xv + xv;

            // Chebyshev chain T0..T15, packed pairwise to f16
            union { pk2 p[8]; v16h v; } a;
            float tp = 1.0f;   // T0
            float tc = xv;     // T1
            a.p[0] = __builtin_amdgcn_cvt_pkrtz(tp, tc);
            #pragma unroll
            for (int j = 1; j < 8; ++j) {
                float t2 = __builtin_fmaf(twox, tc, -tp);  // T_{2j}
                float t3 = __builtin_fmaf(twox, t2, -tc);  // T_{2j+1}
                a.p[j] = __builtin_amdgcn_cvt_pkrtz(t2, t3);
                tp = t2; tc = t3;
            }

            // B fragments for N-tile 0 (o 0..15) and N-tile 1 (o 16..31)
            const v16h b0 = *(const v16h*)&ldsK[(size_t)(c * 2 + 0) * 512 + (size_t)lane * 16];
            const v16h b1 = *(const v16h*)&ldsK[(size_t)(c * 2 + 1) * 512 + (size_t)lane * 16];

            acc0 = __builtin_amdgcn_wmma_f32_16x16x32_f16(
                false, a.v, false, b0, (short)0, acc0, false, false);
            acc1 = __builtin_amdgcn_wmma_f32_16x16x32_f16(
                false, a.v, false, b1, (short)0, acc1, false, false);
        }

        // C/D f32 16x16 layout: VGPR r, lanes 0-15 -> M=r, lanes 16-31 -> M=8+r
        float* __restrict__ orow = out + (size_t)r0 * O_OUT;
        const int mbase = g * 8;
        const int ncol  = lane & 15;
        #pragma unroll
        for (int r = 0; r < 8; ++r) {
            orow[(size_t)(mbase + r) * O_OUT + ncol]      = acc0[r];
            orow[(size_t)(mbase + r) * O_OUT + 16 + ncol] = acc1[r];
        }
    }
}

extern "C" void kernel_launch(void* const* d_in, const int* in_sizes, int n_in,
                              void* d_out, int out_size, void* d_ws, size_t ws_size,
                              hipStream_t stream) {
    const float* x      = (const float*)d_in[0];
    const float* kernel = (const float*)d_in[1];
    float* out          = (float*)d_out;
    hipLaunchKernelGGL(kan_cheb_wmma, dim3(GRID_B), dim3(THREADS), 0, stream,
                       x, kernel, out);
}
